// EncoderBlock_36687610642811
// MI455X (gfx1250) — matrix operand order repaired and checked by reading, not compiled, not run on hardware
//
#include <hip/hip_runtime.h>
#include <hip/hip_bf16.h>

typedef __bf16 bf16_t;
typedef __attribute__((ext_vector_type(8)))  __bf16 v8bf;
typedef __attribute__((ext_vector_type(16))) __bf16 v16bf;
typedef __attribute__((ext_vector_type(8)))  float  v8f;

union AF { v16bf v; v8bf h[2]; };

__device__ __forceinline__ v8f wmma_bf16(const AF& a, const AF& b, v8f c) {
    // D = A(16x32 bf16) x B(32x16 bf16) + C(16x16 f32)
    return __builtin_amdgcn_wmma_f32_16x16x32_bf16(
        false, a.v, false, b.v, (short)0, c, false, false);
}

// 16-lane butterfly reductions via DPP16 row_xmask (no LDS traffic):
// dpp_ctrl 0x160|mask = ROW_XMASK, lanes swap with lane^mask inside 16-lane row.
__device__ __forceinline__ float dppx_f32(float v, const int ctrl_is_inlined_below) {
    return v; // (unused helper placeholder; literal ctrls required by builtin)
}
__device__ __forceinline__ float rmax16(float v) {
    v = fmaxf(v, __int_as_float(__builtin_amdgcn_update_dpp(
            0, __float_as_int(v), 0x161, 0xf, 0xf, true)));
    v = fmaxf(v, __int_as_float(__builtin_amdgcn_update_dpp(
            0, __float_as_int(v), 0x162, 0xf, 0xf, true)));
    v = fmaxf(v, __int_as_float(__builtin_amdgcn_update_dpp(
            0, __float_as_int(v), 0x164, 0xf, 0xf, true)));
    v = fmaxf(v, __int_as_float(__builtin_amdgcn_update_dpp(
            0, __float_as_int(v), 0x168, 0xf, 0xf, true)));
    return v;
}
__device__ __forceinline__ float rsum16(float v) {
    v += __int_as_float(__builtin_amdgcn_update_dpp(
            0, __float_as_int(v), 0x161, 0xf, 0xf, true));
    v += __int_as_float(__builtin_amdgcn_update_dpp(
            0, __float_as_int(v), 0x162, 0xf, 0xf, true));
    v += __int_as_float(__builtin_amdgcn_update_dpp(
            0, __float_as_int(v), 0x164, 0xf, 0xf, true));
    v += __int_as_float(__builtin_amdgcn_update_dpp(
            0, __float_as_int(v), 0x168, 0xf, 0xf, true));
    return v;
}

// ---------------------------------------------------------------------------
// One-time: fp32 W[K][N] -> bf16 Wt[N][K]  (transpose so B-fragments are
// contiguous 16-byte runs along K).
// ---------------------------------------------------------------------------
__global__ void transpose_f32_to_bf16(const float* __restrict__ W,
                                      bf16_t* __restrict__ Wt, int K, int N) {
    long idx = (long)blockIdx.x * blockDim.x + threadIdx.x;
    if (idx >= (long)K * N) return;
    int k = (int)(idx / N), n = (int)(idx % N);
    Wt[(size_t)n * K + k] = (bf16_t)W[idx];
}

// ---------------------------------------------------------------------------
// Custom LayerNorm (scalar alpha/beta, unbiased n-1 var, eps added to std),
// output bf16 for the WMMA GEMMs. One block per row of 1024.
// ---------------------------------------------------------------------------
__global__ __launch_bounds__(256) void layernorm_to_bf16(
    const float* __restrict__ x, bf16_t* __restrict__ out,
    const float* __restrict__ alpha, const float* __restrict__ beta, int d) {
    __shared__ float sred[16];
    const int row = blockIdx.x;
    const float* xr = x + (size_t)row * d;
    float s = 0.f, sq = 0.f;
    for (int i = threadIdx.x; i < d; i += 256) { float t = xr[i]; s += t; sq += t * t; }
    for (int off = 16; off >= 1; off >>= 1) {
        s  += __shfl_xor(s,  off, 32);
        sq += __shfl_xor(sq, off, 32);
    }
    const int lane = threadIdx.x & 31, w = threadIdx.x >> 5;
    if (lane == 0) { sred[w] = s; sred[8 + w] = sq; }
    __syncthreads();
    s = 0.f; sq = 0.f;
    for (int i = 0; i < 8; ++i) { s += sred[i]; sq += sred[8 + i]; }
    float mean = s / (float)d;
    float var  = (sq - mean * s) / (float)(d - 1);   // unbiased
    float stdv = sqrtf(fmaxf(var, 0.f));
    float scale = alpha[0] / (stdv + 1e-6f);
    float shift = beta[0];
    for (int i = threadIdx.x; i < d; i += 256)
        out[(size_t)row * d + i] = (bf16_t)((xr[i] - mean) * scale + shift);
}

// ---------------------------------------------------------------------------
// Tiled WMMA GEMM: C[M,N] = A[M,K](bf16,rm) x Bt[N,K](bf16,rm)^T + bias[N]
// Block: 256 thr (8 waves, 2x4), tile 128x128, K-step 32 staged in LDS.
// Register-prefetch double pump: next tile's global loads overlap the WMMAs.
// EPI 0: f32 out (+ optional residual)
// EPI 1: bf16 out, head-split  [b,h,s,dk]   (Q,K proj)
// EPI 2: bf16 out, transposed  [b,h,dk,s]   (V proj -> V^T for PV WMMA)
// EPI 3: bf16 out, ReLU        [m,n]        (FFN layer 1)
// ---------------------------------------------------------------------------
#define LDS_STRIDE 40   // 32 + 8 bf16 pad

template <int EPI>
__global__ __launch_bounds__(256) void gemm_bf16(
    const bf16_t* __restrict__ A, const bf16_t* __restrict__ Bt,
    const float* __restrict__ bias, const float* __restrict__ res,
    float* __restrict__ outF, bf16_t* __restrict__ outB,
    int M, int N, int K, int S) {
    __shared__ bf16_t As[128 * LDS_STRIDE];
    __shared__ bf16_t Bs[128 * LDS_STRIDE];

    const int tid  = threadIdx.x;
    const int lane = tid & 31;
    const int w    = tid >> 5;
    const int wm   = w & 1;          // 2 waves along M
    const int wn   = w >> 1;         // 4 waves along N
    const int m0   = blockIdx.y * 128;
    const int n0   = blockIdx.x * 128;
    const int lr   = lane & 15;
    const int hs   = lane >> 4;      // 0: lanes 0-15, 1: lanes 16-31

    // each thread copies 2 16B chunks per tile: chunk c -> row c/4, col (c%4)*8
    const int r0 = tid >> 2,          o0 = (tid & 3) * 8;
    const int r1 = (tid + 256) >> 2,  o1 = (tid & 3) * 8;  // (tid+256)&3 == tid&3

    v8f acc[4][2];
    for (int i = 0; i < 4; ++i)
        for (int j = 0; j < 2; ++j)
            for (int c = 0; c < 8; ++c) acc[i][j][c] = 0.f;

    // prologue: fetch tile kt=0 and commit to LDS
    v8bf pa0 = *(const v8bf*)&A [(size_t)(m0 + r0) * K + o0];
    v8bf pa1 = *(const v8bf*)&A [(size_t)(m0 + r1) * K + o1];
    v8bf pb0 = *(const v8bf*)&Bt[(size_t)(n0 + r0) * K + o0];
    v8bf pb1 = *(const v8bf*)&Bt[(size_t)(n0 + r1) * K + o1];
    *(v8bf*)&As[r0 * LDS_STRIDE + o0] = pa0;
    *(v8bf*)&As[r1 * LDS_STRIDE + o1] = pa1;
    *(v8bf*)&Bs[r0 * LDS_STRIDE + o0] = pb0;
    *(v8bf*)&Bs[r1 * LDS_STRIDE + o1] = pb1;

    for (int kt = 0; kt < K; kt += 32) {
        __syncthreads();               // LDS tile for kt is visible
        const bool more = (kt + 32) < K;
        if (more) {                    // prefetch next tile; overlaps the WMMAs
            pa0 = *(const v8bf*)&A [(size_t)(m0 + r0) * K + kt + 32 + o0];
            pa1 = *(const v8bf*)&A [(size_t)(m0 + r1) * K + kt + 32 + o1];
            pb0 = *(const v8bf*)&Bt[(size_t)(n0 + r0) * K + kt + 32 + o0];
            pb1 = *(const v8bf*)&Bt[(size_t)(n0 + r1) * K + kt + 32 + o1];
        }

        AF a[4], b[2];
        for (int i = 0; i < 4; ++i) {   // A frag: row=lane(16), k {0-7,16-23}/{8-15,24-31}
            const bf16_t* p = &As[(wm * 64 + i * 16 + lr) * LDS_STRIDE];
            a[i].h[0] = *(const v8bf*)&p[hs * 8];
            a[i].h[1] = *(const v8bf*)&p[16 + hs * 8];
        }
        for (int j = 0; j < 2; ++j) {   // B frag: col=lane(16), k {0-7,16-23}/{8-15,24-31}
            const bf16_t* p = &Bs[(wn * 32 + j * 16 + lr) * LDS_STRIDE];
            b[j].h[0] = *(const v8bf*)&p[hs * 8];
            b[j].h[1] = *(const v8bf*)&p[16 + hs * 8];
        }
        for (int i = 0; i < 4; ++i)
            for (int j = 0; j < 2; ++j)
                acc[i][j] = wmma_bf16(a[i], b[j], acc[i][j]);

        __syncthreads();               // everyone done reading the tile
        if (more) {                    // commit prefetched tile
            *(v8bf*)&As[r0 * LDS_STRIDE + o0] = pa0;
            *(v8bf*)&As[r1 * LDS_STRIDE + o1] = pa1;
            *(v8bf*)&Bs[r0 * LDS_STRIDE + o0] = pb0;
            *(v8bf*)&Bs[r1 * LDS_STRIDE + o1] = pb1;
        }
    }

    // epilogue: D frag VGPR c -> row (c | c+8), col = lane&15
    for (int i = 0; i < 4; ++i)
        for (int j = 0; j < 2; ++j) {
            const int n = n0 + wn * 32 + j * 16 + lr;
            const float bv = bias[n];
            for (int c = 0; c < 8; ++c) {
                const int m = m0 + wm * 64 + i * 16 + c + hs * 8;
                float val = acc[i][j][c] + bv;
                if (EPI == 0) {
                    float r = res ? res[(size_t)m * N + n] : 0.f;
                    outF[(size_t)m * N + n] = val + r;
                } else if (EPI == 1) {
                    int bb = m / S, ss = m % S, hd = n >> 6, dkk = n & 63;
                    outB[(((size_t)bb * 16 + hd) * S + ss) * 64 + dkk] = (bf16_t)val;
                } else if (EPI == 2) {
                    int bb = m / S, ss = m % S, hd = n >> 6, dkk = n & 63;
                    outB[(((size_t)bb * 16 + hd) * 64 + dkk) * S + ss] = (bf16_t)val;
                } else {
                    outB[(size_t)m * N + n] = (bf16_t)fmaxf(val, 0.f);
                }
            }
        }
}

// ---------------------------------------------------------------------------
// Flash attention: block = 4 waves x 16 q-rows = 64 q-rows per (b,h) tile.
// q,k: [B*h,S,64] bf16 ; vT: [B*h,64,S] bf16 ; ctx out: [B,S,1024] bf16.
// Streaming softmax over 32-key steps; QK^T and PV both via WMMA; row
// statistics via DPP row_xmask butterflies (VALU-only, no LDS).
// ---------------------------------------------------------------------------
__global__ __launch_bounds__(128) void attention_kernel(
    const bf16_t* __restrict__ q, const bf16_t* __restrict__ k,
    const bf16_t* __restrict__ vT, const int* __restrict__ mask,
    bf16_t* __restrict__ ctx, int S) {
    __shared__ bf16_t pbuf[4][16 * 32];   // per-wave P staging (D-layout -> A-layout)

    const int lane = threadIdx.x & 31;
    const int w    = threadIdx.x >> 5;
    const int lr   = lane & 15;
    const int hs   = lane >> 4;
    const int nq   = S / 64;
    const int bh   = blockIdx.x / nq;
    const int qt   = blockIdx.x % nq;
    const int b    = bh >> 4;
    const int head = bh & 15;
    const int q0   = qt * 64 + w * 16;

    const bf16_t* qb = q  + (size_t)bh * S * 64;
    const bf16_t* kb = k  + (size_t)bh * S * 64;
    const bf16_t* vb = vT + (size_t)bh * 64 * S;
    const int*    mb = mask + (size_t)b * S;

    AF qa[2];   // Q rows q0..q0+15, dk chunks 0-31 / 32-63 in A-layout
    {
        const bf16_t* p = qb + (size_t)(q0 + lr) * 64;
        for (int c = 0; c < 2; ++c) {
            qa[c].h[0] = *(const v8bf*)&p[c * 32 + hs * 8];
            qa[c].h[1] = *(const v8bf*)&p[c * 32 + 16 + hs * 8];
        }
    }

    float mrow[8], lrow[8];
    v8f o[4];
    for (int i = 0; i < 8; ++i) { mrow[i] = -3.0e38f; lrow[i] = 0.f; }
    for (int t = 0; t < 4; ++t)
        for (int c = 0; c < 8; ++c) o[t][c] = 0.f;

    for (int jt = 0; jt < S; jt += 32) {
        v8f s0 = {}, s1 = {};
        for (int c = 0; c < 2; ++c) {   // dk chunks; K rows serve as B-frag cols
            AF kb0, kb1;
            const bf16_t* p0 = kb + (size_t)(jt + lr) * 64 + c * 32 + hs * 8;
            kb0.h[0] = *(const v8bf*)&p0[0];
            kb0.h[1] = *(const v8bf*)&p0[16];
            const bf16_t* p1 = kb + (size_t)(jt + 16 + lr) * 64 + c * 32 + hs * 8;
            kb1.h[0] = *(const v8bf*)&p1[0];
            kb1.h[1] = *(const v8bf*)&p1[16];
            s0 = wmma_bf16(qa[c], kb0, s0);
            s1 = wmma_bf16(qa[c], kb1, s1);
        }
        const float madd0 = (mb[jt + lr] == 0)      ? -1e9f : 0.f;
        const float madd1 = (mb[jt + 16 + lr] == 0) ? -1e9f : 0.f;

        float alf[8];
        for (int i = 0; i < 8; ++i) {
            float a0 = s0[i] * 0.125f + madd0;   // 1/sqrt(64)
            float a1 = s1[i] * 0.125f + madd1;
            float tm = rmax16(fmaxf(a0, a1));
            float nm = fmaxf(mrow[i], tm);
            float al = __expf(mrow[i] - nm);
            float p0 = __expf(a0 - nm);
            float p1 = __expf(a1 - nm);
            s0[i] = p0; s1[i] = p1;
            lrow[i] = lrow[i] * al + rsum16(p0 + p1);
            mrow[i] = nm;
            alf[i] = al;
        }
        for (int t = 0; t < 4; ++t)
            for (int i = 0; i < 8; ++i) o[t][i] *= alf[i];

        // P (D-layout f32) -> LDS 16x32 bf16 -> reload in A-layout
        bf16_t* pb = pbuf[w];
        for (int i = 0; i < 8; ++i) {
            int r = i + hs * 8;
            pb[r * 32 + lr]      = (bf16_t)s0[i];
            pb[r * 32 + 16 + lr] = (bf16_t)s1[i];
        }
        asm volatile("s_wait_dscnt 0" ::: "memory");
        AF pa;
        {
            const bf16_t* p = &pb[lr * 32];
            pa.h[0] = *(const v8bf*)&p[hs * 8];
            pa.h[1] = *(const v8bf*)&p[16 + hs * 8];
        }
        for (int t = 0; t < 4; ++t) {   // PV: B-frag cols = dk, k = key (V^T rows)
            AF vf;
            const bf16_t* p = vb + (size_t)(t * 16 + lr) * S + jt + hs * 8;
            vf.h[0] = *(const v8bf*)&p[0];
            vf.h[1] = *(const v8bf*)&p[16];
            o[t] = wmma_bf16(pa, vf, o[t]);
        }
    }

    for (int i = 0; i < 8; ++i) lrow[i] = 1.f / lrow[i];
    for (int t = 0; t < 4; ++t)
        for (int i = 0; i < 8; ++i) {
            int row = q0 + i + hs * 8;
            int col = head * 64 + t * 16 + lr;
            ctx[((size_t)b * S + row) * 1024 + col] = (bf16_t)(o[t][i] * lrow[i]);
        }
}

// ---------------------------------------------------------------------------
extern "C" void kernel_launch(void* const* d_in, const int* in_sizes, int n_in,
                              void* d_out, int out_size, void* d_ws, size_t ws_size,
                              hipStream_t stream) {
    (void)in_sizes; (void)n_in; (void)out_size; (void)ws_size;
    const int B = 2, S = 2048, d = 1024, dff = 4096, M = B * S;

    const float* x      = (const float*)d_in[0];
    const float* wq     = (const float*)d_in[1];
    const float* bq     = (const float*)d_in[2];
    const float* wk     = (const float*)d_in[3];
    const float* bk     = (const float*)d_in[4];
    const float* wv     = (const float*)d_in[5];
    const float* bv     = (const float*)d_in[6];
    const float* wo     = (const float*)d_in[7];
    const float* bo     = (const float*)d_in[8];
    const float* w1     = (const float*)d_in[9];
    const float* b1     = (const float*)d_in[10];
    const float* w2     = (const float*)d_in[11];
    const float* b2     = (const float*)d_in[12];
    const float* alpha1 = (const float*)d_in[13];
    const float* beta1  = (const float*)d_in[14];
    const float* alpha2 = (const float*)d_in[15];
    const float* beta2  = (const float*)d_in[16];
    const int*   smask  = (const int*)d_in[17];
    float*       out    = (float*)d_out;

    char* ws = (char*)d_ws;
    auto alloc = [&](size_t bytes) -> void* {
        void* p = ws; ws += (bytes + 255) & ~(size_t)255; return p;
    };
    bf16_t* WqT = (bf16_t*)alloc((size_t)d * d * 2);
    bf16_t* WkT = (bf16_t*)alloc((size_t)d * d * 2);
    bf16_t* WvT = (bf16_t*)alloc((size_t)d * d * 2);
    bf16_t* WoT = (bf16_t*)alloc((size_t)d * d * 2);
    bf16_t* W1T = (bf16_t*)alloc((size_t)d * dff * 2);
    bf16_t* W2T = (bf16_t*)alloc((size_t)dff * d * 2);
    bf16_t* xn  = (bf16_t*)alloc((size_t)M * d * 2);
    bf16_t* qb  = (bf16_t*)alloc((size_t)M * d * 2);
    bf16_t* kbf = (bf16_t*)alloc((size_t)M * d * 2);
    bf16_t* vT  = (bf16_t*)alloc((size_t)M * d * 2);
    bf16_t* ctx = (bf16_t*)alloc((size_t)M * d * 2);
    float*  x2  = (float*) alloc((size_t)M * d * 4);
    bf16_t* xn2 = (bf16_t*)alloc((size_t)M * d * 2);
    bf16_t* hh  = (bf16_t*)alloc((size_t)M * dff * 2);

    // 1) weights -> bf16, transposed
    int ndd = (d * d + 255) / 256, ndf = (d * dff + 255) / 256;
    transpose_f32_to_bf16<<<ndd, 256, 0, stream>>>(wq, WqT, d, d);
    transpose_f32_to_bf16<<<ndd, 256, 0, stream>>>(wk, WkT, d, d);
    transpose_f32_to_bf16<<<ndd, 256, 0, stream>>>(wv, WvT, d, d);
    transpose_f32_to_bf16<<<ndd, 256, 0, stream>>>(wo, WoT, d, d);
    transpose_f32_to_bf16<<<ndf, 256, 0, stream>>>(w1, W1T, d, dff);
    transpose_f32_to_bf16<<<ndf, 256, 0, stream>>>(w2, W2T, dff, d);

    // 2) LN1 -> bf16
    layernorm_to_bf16<<<M, 256, 0, stream>>>(x, xn, alpha1, beta1, d);

    // 3) Q,K,V projections (WMMA), V written transposed
    dim3 gdd(d / 128, M / 128), gdf(dff / 128, M / 128);
    gemm_bf16<1><<<gdd, 256, 0, stream>>>(xn, WqT, bq, nullptr, nullptr, qb,  M, d, d, S);
    gemm_bf16<1><<<gdd, 256, 0, stream>>>(xn, WkT, bk, nullptr, nullptr, kbf, M, d, d, S);
    gemm_bf16<2><<<gdd, 256, 0, stream>>>(xn, WvT, bv, nullptr, nullptr, vT,  M, d, d, S);

    // 4) flash attention -> ctx (head-merged bf16)
    attention_kernel<<<B * 16 * (S / 64), 128, 0, stream>>>(qb, kbf, vT, smask, ctx, S);

    // 5) output projection + residual -> x2 (f32)
    gemm_bf16<0><<<gdd, 256, 0, stream>>>(ctx, WoT, bo, x, x2, nullptr, M, d, d, S);

    // 6) LN2, FFN (ReLU fused), second FFN GEMM + residual -> out
    layernorm_to_bf16<<<M, 256, 0, stream>>>(x2, xn2, alpha2, beta2, d);
    gemm_bf16<3><<<gdf, 256, 0, stream>>>(xn2, W1T, b1, nullptr, nullptr, hh, M, dff, d, S);
    gemm_bf16<0><<<gdd, 256, 0, stream>>>(hh, W2T, b2, x2, out, nullptr, M, d, dff, S);
}
